// STransferEncoder_71562745086229
// MI455X (gfx1250) — matrix operand order repaired and checked by reading, not compiled, not run on hardware
//
#include <hip/hip_runtime.h>

// ---------------------------------------------------------------------------
// CDNA5 (gfx1250) implementation of the STransfer graph-VAE encoder.
// wave32; all dense GEMMs via v_wmma_f32_16x16x32_bf16 with:
//   - weights pre-converted once to bf16, transposed (K-contiguous) so a
//     B-fragment is two 16B contiguous loads,
//   - A tiles staged in LDS as bf16 (converted once per element),
//   - fully unrolled K loops (dense WMMA stream).
// Graph phase: one wave32 per edge (lane == feature), L2-resident atomics.
// ---------------------------------------------------------------------------

#define NODES 100000
#define EDGES 2400000
#define DIN   256

typedef __attribute__((ext_vector_type(16))) __bf16   v16bf;
typedef __attribute__((ext_vector_type(8)))  float    v8f;
typedef __attribute__((ext_vector_type(4)))  unsigned uint32x4;

union UB16 {
  v16bf    v;
  unsigned u[8];
  uint32x4 q[2];
};

__device__ __forceinline__ unsigned short tobf_bits(float f) {
  unsigned u = __builtin_bit_cast(unsigned, f);
  unsigned r = u + 0x7FFFu + ((u >> 16) & 1u);   // round-to-nearest-even
  return (unsigned short)(r >> 16);
}

__device__ __forceinline__ unsigned packbf(float a, float b) {
  return (unsigned)tobf_bits(a) | ((unsigned)tobf_bits(b) << 16);
}

__device__ __forceinline__ float eluf(float v) {
  return v > 0.f ? v : (__expf(v) - 1.f);
}

// A-fragment (16x32 bf16) from LDS bf16 tile, row-major [16][ldk].
// Lane group 0 (lanes 0-15): K {0..15}; group 1: K {16..31}.
__device__ __forceinline__ v16bf load_afrag(const unsigned short* lds, int m,
                                            int ldk, int k0, int ka) {
  UB16 r;
  const unsigned short* p = lds + m * ldk + k0 + 2 * ka;   // 4B aligned (2*ka in {0,16})
#pragma unroll
  for (int j = 0; j < 4; ++j) {
    r.u[j]     = *(const unsigned*)(p + 2 * j);
    r.u[j + 4] = *(const unsigned*)(p + 8 + 2 * j);
  }
  return r.v;
}

// B-fragment (32x16 bf16) from transposed bf16 weights Wt[n][K] (K-contiguous).
// Lane group 0: K {0..15}; group 1: K {16..31}. Two 16B loads.
__device__ __forceinline__ v16bf load_bfrag(const unsigned short* Wt, int ncol,
                                            int K, int k0, int kb) {
  UB16 r;
  const uint32x4* p = (const uint32x4*)(Wt + (size_t)ncol * K + k0 + kb);
  r.q[0] = p[0];
  r.q[1] = p[1];
  return r.v;
}

// ---------------------------------------------------------------------------
// small utility kernels
// ---------------------------------------------------------------------------
__global__ void k_zero_stats(float* stat1, float* stat2) {
  int t = threadIdx.x;
  if (t < 128) stat1[t] = 0.f;
  if (t < 64)  stat2[t] = 0.f;
}

// W[k][n] f32 -> Wt[n][k] bf16
__global__ void k_cvtW(const float* __restrict__ W, unsigned short* __restrict__ Wt,
                       int K, int N) {
  int i = blockIdx.x * blockDim.x + threadIdx.x;
  if (i >= K * N) return;
  int k = i / N, n = i - k * N;
  Wt[(size_t)n * K + k] = tobf_bits(W[i]);
}

__global__ void k_deg_init(float* dg, float* dp) {
  int i = blockIdx.x * blockDim.x + threadIdx.x;
  if (i < NODES) { dg[i] = 2.f; dp[i] = 2.f; }  // self-loop fill_value = 2
}

__global__ void k_deg(const int* __restrict__ src, const float* __restrict__ w,
                      float* dg, float* dp) {
  int e = blockIdx.x * blockDim.x + threadIdx.x;
  if (e >= EDGES) return;
  int s = src[e];
  atomicAdd(&dg[s], 1.0f);
  atomicAdd(&dp[s], w[e]);
}

__global__ void k_rsqrt(float* dg, float* dp) {
  int i = blockIdx.x * blockDim.x + threadIdx.x;
  if (i < NODES) { dg[i] = rsqrtf(dg[i]); dp[i] = rsqrtf(dp[i]); }
}

__global__ void k_bnparam(const float* __restrict__ stat, const float* __restrict__ g,
                          const float* __restrict__ be, float* p, int C) {
  int j = threadIdx.x;
  if (j >= C) return;
  float m = stat[j] / (float)NODES;
  float v = stat[C + j] / (float)NODES - m * m;
  float a = g[j] * rsqrtf(v + 1e-3f);
  p[j]     = a;
  p[C + j] = be[j] - a * m;
}

__global__ void k_relu(float* x) {
  int i = blockIdx.x * blockDim.x + threadIdx.x;
  if (i < NODES * 32) x[i] = fmaxf(x[i], 0.f);
}

// out[i,k] = bias[k] + 2*dis[i]^2 * t[i,k]   (self-loop contribution + bias)
__global__ void k_conv_init(const float* __restrict__ t, const float* __restrict__ dis,
                            const float* __restrict__ bias, float* __restrict__ out) {
  int i = blockIdx.x * blockDim.x + threadIdx.x;
  if (i >= NODES * 32) return;
  int node = i >> 5, k = i & 31;
  float d = dis[node];
  out[i] = bias[k] + 2.f * d * d * t[i];
}

// one wave per edge; lane = feature; atomic scatter into L2-resident out
__global__ void k_scatter(const int* __restrict__ src, const int* __restrict__ dst,
                          const float* __restrict__ w, const float* __restrict__ dis,
                          const float* __restrict__ t, float* __restrict__ out) {
  long long gid = (long long)blockIdx.x * blockDim.x + threadIdx.x;
  int e = (int)(gid >> 5), lane = (int)(gid & 31);
  if (e >= EDGES) return;
  int s = src[e], d = dst[e];
  float nw = dis[s] * dis[d] * (w ? w[e] : 1.f);
  atomicAdd(&out[(size_t)d * 32 + lane], nw * t[(size_t)s * 32 + lane]);
}

// fused dual scatter (mu & logvar share the edge norm)
__global__ void k_scatter2(const int* __restrict__ src, const int* __restrict__ dst,
                           const float* __restrict__ w, const float* __restrict__ dis,
                           const float* __restrict__ t2, const float* __restrict__ t3,
                           float* __restrict__ o2, float* __restrict__ o3) {
  long long gid = (long long)blockIdx.x * blockDim.x + threadIdx.x;
  int e = (int)(gid >> 5), lane = (int)(gid & 31);
  if (e >= EDGES) return;
  int s = src[e], d = dst[e];
  float nw = dis[s] * dis[d] * (w ? w[e] : 1.f);
  size_t si = (size_t)s * 32 + lane, di = (size_t)d * 32 + lane;
  atomicAdd(&o2[di], nw * t2[si]);
  atomicAdd(&o3[di], nw * t3[si]);
}

// ---------------------------------------------------------------------------
// GEMM1: h1 = x[100k,256] @ W1[256,64] + b1 ; per-column BN stats.
// 4 waves / block; wave w owns N-tile w; K loop fully unrolled (8 WMMA).
// ---------------------------------------------------------------------------
__global__ void k_gemm1(const float* __restrict__ x, const unsigned short* __restrict__ W1t,
                        const float* __restrict__ b1, float* __restrict__ h1,
                        float* __restrict__ stat, int ntiles) {
  __shared__ unsigned short As[16 * 256];
  int tile = blockIdx.x, tid = threadIdx.x;
  int wave = tid >> 5, lane = tid & 31;

  const float4* xr = (const float4*)(x + (size_t)tile * 16 * 256);
  if (tile + 1 < ntiles)                       // stream next M-tile into cache
    __builtin_prefetch(x + (size_t)(tile + 1) * 16 * 256 + tid * 32, 0, 1);
  for (int i = tid; i < 1024; i += 128) {
    float4 v = xr[i];
    ((unsigned*)As)[2 * i]     = packbf(v.x, v.y);
    ((unsigned*)As)[2 * i + 1] = packbf(v.z, v.w);
  }
  __syncthreads();

  int m = lane & 15;
  int ncol = wave * 16 + m;
  int ka = (lane >> 4) * 8, kb = (lane >> 4) * 16;
  v8f acc = {};
#pragma unroll
  for (int k0 = 0; k0 < 256; k0 += 32) {
    v16bf a = load_afrag(As, m, 256, k0, ka);
    v16bf b = load_bfrag(W1t, ncol, 256, k0, kb);
    acc = __builtin_amdgcn_wmma_f32_16x16x32_bf16(false, a, false, b,
                                                  (short)0, acc, false, false);
  }
  float bias = b1[ncol];
  float s = 0.f, ss = 0.f;
#pragma unroll
  for (int r = 0; r < 8; ++r) {
    int row = tile * 16 + r + 8 * (lane >> 4);
    float v = acc[r] + bias;
    h1[(size_t)row * 64 + ncol] = v;
    s += v; ss += v * v;
  }
  s  += __shfl_xor(s, 16, 32);
  ss += __shfl_xor(ss, 16, 32);
  if (lane < 16) {
    atomicAdd(&stat[ncol], s);
    atomicAdd(&stat[64 + ncol], ss);
  }
}

// ---------------------------------------------------------------------------
// GEMM2: h2 = ELU(a1*h1+c1) @ W2[64,32] + b2 ; BN2 stats. 2 waves / block.
// ---------------------------------------------------------------------------
__global__ void k_gemm2(const float* __restrict__ h1, const unsigned short* __restrict__ W2t,
                        const float* __restrict__ b2, const float* __restrict__ p1,
                        float* __restrict__ h2, float* __restrict__ stat) {
  __shared__ unsigned short As[16 * 64];
  int tile = blockIdx.x, tid = threadIdx.x;
  int wave = tid >> 5, lane = tid & 31;

  const float* srcp = h1 + (size_t)tile * 16 * 64;
  for (int i = tid; i < 512; i += 64) {          // 512 bf16-pairs
    int e = 2 * i, c0 = e & 63;
    float v0 = eluf(srcp[e]     * p1[c0]     + p1[64 + c0]);
    float v1 = eluf(srcp[e + 1] * p1[c0 + 1] + p1[64 + c0 + 1]);
    ((unsigned*)As)[i] = packbf(v0, v1);
  }
  __syncthreads();

  int m = lane & 15;
  int ncol = wave * 16 + m;
  int ka = (lane >> 4) * 8, kb = (lane >> 4) * 16;
  v8f acc = {};
#pragma unroll
  for (int k0 = 0; k0 < 64; k0 += 32) {
    v16bf a = load_afrag(As, m, 64, k0, ka);
    v16bf b = load_bfrag(W2t, ncol, 64, k0, kb);
    acc = __builtin_amdgcn_wmma_f32_16x16x32_bf16(false, a, false, b,
                                                  (short)0, acc, false, false);
  }
  float bias = b2[ncol];
  float s = 0.f, ss = 0.f;
#pragma unroll
  for (int r = 0; r < 8; ++r) {
    int row = tile * 16 + r + 8 * (lane >> 4);
    float v = acc[r] + bias;
    h2[(size_t)row * 32 + ncol] = v;
    s += v; ss += v * v;
  }
  s  += __shfl_xor(s, 16, 32);
  ss += __shfl_xor(ss, 16, 32);
  if (lane < 16) {
    atomicAdd(&stat[ncol], s);
    atomicAdd(&stat[32 + ncol], ss);
  }
}

// ---------------------------------------------------------------------------
// GEMM3: t1 = ELU(a2*h2+c2) @ Wg1[32,32]. K=32, single WMMA, 2 waves.
// ---------------------------------------------------------------------------
__global__ void k_gemm3(const float* __restrict__ h2, const unsigned short* __restrict__ Wg1t,
                        const float* __restrict__ p2, float* __restrict__ t1) {
  __shared__ unsigned short As[16 * 32];
  int tile = blockIdx.x, tid = threadIdx.x;
  int wave = tid >> 5, lane = tid & 31;

  const float* srcp = h2 + (size_t)tile * 16 * 32;
  for (int i = tid; i < 256; i += 64) {
    int e = 2 * i, c0 = e & 31;
    float v0 = eluf(srcp[e]     * p2[c0]     + p2[32 + c0]);
    float v1 = eluf(srcp[e + 1] * p2[c0 + 1] + p2[32 + c0 + 1]);
    ((unsigned*)As)[i] = packbf(v0, v1);
  }
  __syncthreads();

  int m = lane & 15;
  int ncol = wave * 16 + m;
  int ka = (lane >> 4) * 8, kb = (lane >> 4) * 16;
  v16bf a = load_afrag(As, m, 32, 0, ka);
  v16bf b = load_bfrag(Wg1t, ncol, 32, 0, kb);
  v8f acc = {};
  acc = __builtin_amdgcn_wmma_f32_16x16x32_bf16(false, a, false, b,
                                                (short)0, acc, false, false);
#pragma unroll
  for (int r = 0; r < 8; ++r) {
    int row = tile * 16 + r + 8 * (lane >> 4);
    t1[(size_t)row * 32 + ncol] = acc[r];
  }
}

// ---------------------------------------------------------------------------
// Dual GEMM: t2 = A @ Wg2, t3 = A @ Wg3 (shared LDS A-tile). 4 waves / block.
// ---------------------------------------------------------------------------
__global__ void k_gemm_dual(const float* __restrict__ A,
                            const unsigned short* __restrict__ W2t,
                            const unsigned short* __restrict__ W3t,
                            float* __restrict__ out2, float* __restrict__ out3) {
  __shared__ unsigned short As[16 * 32];
  int tile = blockIdx.x, tid = threadIdx.x;
  int wave = tid >> 5, lane = tid & 31;

  const float2* srcp = (const float2*)(A + (size_t)tile * 16 * 32);
  for (int i = tid; i < 256; i += 128) {
    float2 v = srcp[i];
    ((unsigned*)As)[i] = packbf(v.x, v.y);
  }
  __syncthreads();

  const unsigned short* Wt = (wave < 2) ? W2t : W3t;
  float* out               = (wave < 2) ? out2 : out3;
  int m = lane & 15;
  int ncol = (wave & 1) * 16 + m;
  int ka = (lane >> 4) * 8, kb = (lane >> 4) * 16;
  v16bf a = load_afrag(As, m, 32, 0, ka);
  v16bf b = load_bfrag(Wt, ncol, 32, 0, kb);
  v8f acc = {};
  acc = __builtin_amdgcn_wmma_f32_16x16x32_bf16(false, a, false, b,
                                                (short)0, acc, false, false);
#pragma unroll
  for (int r = 0; r < 8; ++r) {
    int row = tile * 16 + r + 8 * (lane >> 4);
    out[(size_t)row * 32 + ncol] = acc[r];
  }
}

// ---------------------------------------------------------------------------
// Final: reparameterize + 2-way attention softmax. One wave per node.
// ---------------------------------------------------------------------------
__global__ void k_att(const float* __restrict__ fp, const float* __restrict__ fa,
                      const float* __restrict__ mup, const float* __restrict__ lvp,
                      const float* __restrict__ mua, const float* __restrict__ lva,
                      const float* __restrict__ np, const float* __restrict__ na,
                      const float* __restrict__ Watt, const float* __restrict__ batt,
                      float* __restrict__ out) {
  long long gid = (long long)blockIdx.x * blockDim.x + threadIdx.x;
  int node = (int)(gid >> 5), lane = (int)(gid & 31);
  if (node >= NODES) return;
  size_t idx = (size_t)node * 32 + lane;
  float wk = Watt[lane], b0 = batt[0];

  float s0 = fp[idx], s1 = fa[idx];
  float l0 = s0 * wk, l1 = s1 * wk;
#pragma unroll
  for (int o = 16; o; o >>= 1) { l0 += __shfl_xor(l0, o, 32); l1 += __shfl_xor(l1, o, 32); }
  l0 += b0; l1 += b0;
  float mx = fmaxf(l0, l1);
  float e0 = __expf(l0 - mx), e1 = __expf(l1 - mx);
  float inv = 1.f / (e0 + e1);
  out[(size_t)node * 64 + lane] = (e0 * s0 + e1 * s1) * inv;

  float zp = np[idx] * __expf(lvp[idx]) + mup[idx];
  float za = na[idx] * __expf(lva[idx]) + mua[idx];
  float m0 = zp * wk, m1 = za * wk;
#pragma unroll
  for (int o = 16; o; o >>= 1) { m0 += __shfl_xor(m0, o, 32); m1 += __shfl_xor(m1, o, 32); }
  m0 += b0; m1 += b0;
  mx = fmaxf(m0, m1);
  e0 = __expf(m0 - mx); e1 = __expf(m1 - mx);
  inv = 1.f / (e0 + e1);
  out[(size_t)node * 64 + 32 + lane] = (e0 * zp + e1 * za) * inv;
}

// ---------------------------------------------------------------------------
extern "C" void kernel_launch(void* const* d_in, const int* in_sizes, int n_in,
                              void* d_out, int out_size, void* d_ws, size_t ws_size,
                              hipStream_t stream) {
  const float* x    = (const float*)d_in[0];
  const int*   eidx = (const int*)d_in[1];
  const float* ppmi = (const float*)d_in[2];
  const float* np   = (const float*)d_in[3];
  const float* na   = (const float*)d_in[4];
  const float* W1   = (const float*)d_in[5];
  const float* b1   = (const float*)d_in[6];
  const float* g1   = (const float*)d_in[7];
  const float* be1  = (const float*)d_in[8];
  const float* W2   = (const float*)d_in[9];
  const float* b2   = (const float*)d_in[10];
  const float* g2   = (const float*)d_in[11];
  const float* be2  = (const float*)d_in[12];
  const float* Wg1  = (const float*)d_in[13];
  const float* bg1  = (const float*)d_in[14];
  const float* Wg2  = (const float*)d_in[15];
  const float* bg2  = (const float*)d_in[16];
  const float* Wg3  = (const float*)d_in[17];
  const float* bg3  = (const float*)d_in[18];
  const float* Watt = (const float*)d_in[19];
  const float* batt = (const float*)d_in[20];
  float* out = (float*)d_out;

  const int* src = eidx;           // row
  const int* dst = eidx + EDGES;   // col

  // workspace layout (float units) with buffer reuse
  float* ws = (float*)d_ws;
  size_t o = 0;
  float* bufA = ws + o; o += (size_t)NODES * 64;  // h1 -> later {t2p, t3p}
  float* bufB = ws + o; o += (size_t)NODES * 32;  // h2 -> later t2a
  float* bufC = ws + o; o += (size_t)NODES * 32;  // t1 -> later t3a
  float* fp   = ws + o; o += (size_t)NODES * 32;
  float* fa   = ws + o; o += (size_t)NODES * 32;
  float* mup  = ws + o; o += (size_t)NODES * 32;
  float* lvp  = ws + o; o += (size_t)NODES * 32;
  float* mua  = ws + o; o += (size_t)NODES * 32;
  float* lva  = ws + o; o += (size_t)NODES * 32;
  float* disg = ws + o; o += NODES;               // deg -> rsqrt in place
  float* disp = ws + o; o += NODES;
  float* stat1 = ws + o; o += 128;
  float* stat2 = ws + o; o += 64;
  float* p1    = ws + o; o += 128;
  float* p2    = ws + o; o += 64;
  unsigned short* W1t  = (unsigned short*)(ws + o); o += (256 * 64) / 2;  // bf16, transposed
  unsigned short* W2t  = (unsigned short*)(ws + o); o += (64 * 32) / 2;
  unsigned short* Wg1t = (unsigned short*)(ws + o); o += (32 * 32) / 2;
  unsigned short* Wg2t = (unsigned short*)(ws + o); o += (32 * 32) / 2;
  unsigned short* Wg3t = (unsigned short*)(ws + o); o += (32 * 32) / 2;

  const int TPB = 256;
  const int nodeBlk = (NODES + TPB - 1) / TPB;
  const int elemBlk = (NODES * 32) / TPB;           // 12500
  const int edgeBlk = (EDGES + TPB - 1) / TPB;
  const int scatBlk = (int)(((long long)EDGES * 32) / TPB);  // 300000
  const int tiles   = NODES / 16;                   // 6250 (exact)

  // weight conversion (bf16, transposed) + degrees + BN stat reset
  k_zero_stats<<<1, 256, 0, stream>>>(stat1, stat2);
  k_cvtW<<<(256 * 64 + TPB - 1) / TPB, TPB, 0, stream>>>(W1, W1t, 256, 64);
  k_cvtW<<<(64 * 32 + TPB - 1) / TPB, TPB, 0, stream>>>(W2, W2t, 64, 32);
  k_cvtW<<<(32 * 32 + TPB - 1) / TPB, TPB, 0, stream>>>(Wg1, Wg1t, 32, 32);
  k_cvtW<<<(32 * 32 + TPB - 1) / TPB, TPB, 0, stream>>>(Wg2, Wg2t, 32, 32);
  k_cvtW<<<(32 * 32 + TPB - 1) / TPB, TPB, 0, stream>>>(Wg3, Wg3t, 32, 32);
  k_deg_init<<<nodeBlk, TPB, 0, stream>>>(disg, disp);
  k_deg<<<edgeBlk, TPB, 0, stream>>>(src, ppmi, disg, disp);
  k_rsqrt<<<nodeBlk, TPB, 0, stream>>>(disg, disp);

  // encoder
  k_gemm1<<<tiles, 128, 0, stream>>>(x, W1t, b1, bufA, stat1, tiles);
  k_bnparam<<<1, 64, 0, stream>>>(stat1, g1, be1, p1, 64);
  k_gemm2<<<tiles, 64, 0, stream>>>(bufA, W2t, b2, p1, bufB, stat2);
  k_bnparam<<<1, 32, 0, stream>>>(stat2, g2, be2, p2, 32);
  k_gemm3<<<tiles, 64, 0, stream>>>(bufB, Wg1t, p2, bufC);   // t1 = emb @ Wg1

  // first conv layer: shared t1, two norms
  k_conv_init<<<elemBlk, TPB, 0, stream>>>(bufC, disp, bg1, fp);
  k_conv_init<<<elemBlk, TPB, 0, stream>>>(bufC, disg, bg1, fa);
  k_scatter<<<scatBlk, TPB, 0, stream>>>(src, dst, ppmi, disp, bufC, fp);
  k_scatter<<<scatBlk, TPB, 0, stream>>>(src, dst, nullptr, disg, bufC, fa);
  k_relu<<<elemBlk, TPB, 0, stream>>>(fp);
  k_relu<<<elemBlk, TPB, 0, stream>>>(fa);

  // second conv layer: dual GEMMs then fused dual scatters
  float* t2p = bufA;
  float* t3p = bufA + (size_t)NODES * 32;
  float* t2a = bufB;
  float* t3a = bufC;
  k_gemm_dual<<<tiles, 128, 0, stream>>>(fp, Wg2t, Wg3t, t2p, t3p);
  k_gemm_dual<<<tiles, 128, 0, stream>>>(fa, Wg2t, Wg3t, t2a, t3a);
  k_conv_init<<<elemBlk, TPB, 0, stream>>>(t2p, disp, bg2, mup);
  k_conv_init<<<elemBlk, TPB, 0, stream>>>(t3p, disp, bg3, lvp);
  k_conv_init<<<elemBlk, TPB, 0, stream>>>(t2a, disg, bg2, mua);
  k_conv_init<<<elemBlk, TPB, 0, stream>>>(t3a, disg, bg3, lva);
  k_scatter2<<<scatBlk, TPB, 0, stream>>>(src, dst, ppmi, disp, t2p, t3p, mup, lvp);
  k_scatter2<<<scatBlk, TPB, 0, stream>>>(src, dst, nullptr, disg, t2a, t3a, mua, lva);

  // reparameterize + attention
  k_att<<<elemBlk, TPB, 0, stream>>>(fp, fa, mup, lvp, mua, lva, np, na, Watt, batt, out);
}
